// ResNetGNN_22789096473213
// MI455X (gfx1250) — matrix operand order repaired and checked by reading, not compiled, not run on hardware
//
#include <hip/hip_runtime.h>
#include <hip/hip_bf16.h>

typedef __attribute__((ext_vector_type(16))) _Float16 v16h;
typedef __attribute__((ext_vector_type(8)))  _Float16 v8h;
typedef __attribute__((ext_vector_type(8)))  float    v8f;

#define NN 20000
#define NE 320000

// ---------------------------------------------------------------- WMMA core
__device__ __forceinline__ int lane_id() { return threadIdx.x & 31; }

__device__ __forceinline__ v8f wmma_acc(v16h a, v16h b, v8f c) {
  // D = A(16x32 f16) * B(32x16 f16) + C(16x16 f32)
  return __builtin_amdgcn_wmma_f32_16x16x32_f16(false, a, false, b, (short)0, c,
                                                false, false);
}

// Hard scheduling fence: nothing may cross (keeps load phase / math phase apart).
__device__ __forceinline__ void sched_fence() {
#if defined(__has_builtin)
#if __has_builtin(__builtin_amdgcn_sched_barrier)
  __builtin_amdgcn_sched_barrier(0);
#endif
#endif
}

// A fragment (16x32, f16) from a row-major buffer (global or LDS), ld in halves.
// Lanes 0-15 -> K {0..7,16..23}, lanes 16-31 -> K {8..15,24..31}; M = lane&15.
__device__ __forceinline__ v16h load_a(const _Float16* p0, int ld, int k0) {
  int l = lane_id();
  const _Float16* p = p0 + (l & 15) * ld + k0 + ((l >> 4) * 8);
  union { v16h v; v8h h[2]; } u;
  u.h[0] = *(const v8h*)p;
  u.h[1] = *(const v8h*)(p + 16);
  return u.v;
}

// B fragment (32x16, f16) from transposed weights Wt[dout][ld] (row-major).
// Lanes 0-15 -> K {0..7,16..23}, lanes 16-31 -> K {8..15,24..31}, N = n0 + lane&15.
__device__ __forceinline__ v16h load_b(const _Float16* wt, int ld, int n0, int k0) {
  int l = lane_id();
  const _Float16* p = wt + (size_t)(n0 + (l & 15)) * ld + k0 + ((l >> 4) * 8);
  union { v16h v; v8h h[2]; } u;
  u.h[0] = *(const v8h*)p;
  u.h[1] = *(const v8h*)(p + 16);
  return u.v;
}

// k-loop with a *binding* load/math phase split: all NF B-fragments are issued
// first (one ds clause, distinct VGPRs), a sched_barrier pins them there, then
// the WMMA chain runs with incremental dscnt waits instead of wait-0 per step.
template <int NF>
__device__ __forceinline__ v8f kloop(const v16h* af, const _Float16* cur, int ldp) {
  v16h bf[NF];
#pragma unroll
  for (int kk = 0; kk < NF; ++kk) bf[kk] = load_b(cur, ldp, 0, kk * 32);
  sched_fence();
  v8f acc = {};
#pragma unroll
  for (int kk = 0; kk < NF; ++kk) acc = wmma_acc(af[kk], bf[kk], acc);
  return acc;
}

// C/D epilogue: bias + optional relu, store f16 row-major (ld halves).
__device__ __forceinline__ void store_tile(const v8f& acc, const float* bias, int n0,
                                           _Float16* out, int ld, bool relu) {
  int l = lane_id();
  int n = n0 + (l & 15);
  float bv = bias[n];
  int mb = (l >> 4) * 8;
#pragma unroll
  for (int r = 0; r < 8; ++r) {
    float v = acc[r] + bv;
    if (relu) v = fmaxf(v, 0.0f);
    out[(mb + r) * ld + n] = (_Float16)v;
  }
}

__device__ __forceinline__ v8h cvt8(const float* p) {
  v8h r;
#pragma unroll
  for (int i = 0; i < 8; ++i) r[i] = (_Float16)p[i];
  return r;
}

// Block-cooperative copy of a 16-column weight panel Wt[n0..n0+15][0..KIN) into LDS
// (row stride ldp halves, padded to stagger banks). 256 threads.
template <int KIN>
__device__ __forceinline__ void load_panel(const _Float16* __restrict__ wt, int n0,
                                           _Float16* pan, int ldp) {
  constexpr int CH = KIN / 8;  // 16-byte chunks per row
#pragma unroll
  for (int c = threadIdx.x; c < 16 * CH; c += 256) {
    int nn = c / CH;
    int k8 = (c - nn * CH) * 8;
    *(v8h*)(pan + nn * ldp + k8) = *(const v8h*)(wt + (size_t)(n0 + nn) * KIN + k8);
  }
}

// One block-cooperative dense layer over this wave's 16-row slab.
template <int KIN, int DOUT>
__device__ __forceinline__ void blayer(const _Float16* aIn, int ldaIn,
                                       const _Float16* __restrict__ wt,
                                       const float* __restrict__ bias,
                                       _Float16* outW, int ldo, bool relu,
                                       _Float16* bpan, int ldp, bool on) {
  constexpr int NF = KIN / 32;
  v16h af[NF];
  if (on) {
#pragma unroll
    for (int kk = 0; kk < NF; ++kk) af[kk] = load_a(aIn, ldaIn, kk * 32);
  }
  load_panel<KIN>(wt, 0, bpan, ldp);
  __syncthreads();
  for (int n0 = 0; n0 < DOUT; n0 += 16) {
    const _Float16* cur = bpan + ((n0 >> 4) & 1) * (16 * ldp);
    _Float16* nxt = bpan + (((n0 >> 4) + 1) & 1) * (16 * ldp);
    if (n0 + 16 < DOUT) load_panel<KIN>(wt, n0 + 16, nxt, ldp);
    if (on) {
      v8f acc = kloop<NF>(af, cur, ldp);
      store_tile(acc, bias, n0, outW, ldo, relu);
    }
    __syncthreads();
  }
}

// ---------------------------------------------------------------- prep kernels

__global__ void wcast_kernel(const float* __restrict__ src, _Float16* __restrict__ dst,
                             int din, int dout, int kpad) {
  int t = blockIdx.x * 256 + threadIdx.x;
  if (t >= dout * kpad) return;
  int o = t / kpad, i = t - o * kpad;
  dst[t] = (i < din) ? (_Float16)src[(size_t)i * dout + o] : (_Float16)0.0f;
}

__global__ void zero_f32(float* p, int n) {
  int t = blockIdx.x * 256 + threadIdx.x;
  if (t < n) p[t] = 0.0f;
}

// node features: [pos(2), vel(2), a0*4, a1*4] padded to 32, f16
__global__ void build_nf0(const float* __restrict__ x, const float* __restrict__ atab,
                          _Float16* __restrict__ nf0, int n) {
  int t = blockIdx.x * 256 + threadIdx.x;
  if (t >= n) return;
  const float* xr = x + (size_t)t * 7;
  int ty = (int)xr[6];
  float a0 = atab[(size_t)ty * 2 + 0], a1 = atab[(size_t)ty * 2 + 1];
  _Float16* o = nf0 + (size_t)t * 32;
  o[0] = (_Float16)xr[0]; o[1] = (_Float16)xr[1];
  o[2] = (_Float16)xr[2]; o[3] = (_Float16)xr[3];
#pragma unroll
  for (int i = 0; i < 4; ++i) o[4 + i] = (_Float16)a0;
#pragma unroll
  for (int i = 0; i < 4; ++i) o[8 + i] = (_Float16)a1;
#pragma unroll
  for (int i = 12; i < 32; ++i) o[i] = (_Float16)0.0f;
}

// edge features: [dpos(2), |dpos|, vi(2), vj(2), a0_i*4] padded to 32, f16 (vnorm==1)
__global__ void build_ef0(const float* __restrict__ x, const float* __restrict__ atab,
                          const int* __restrict__ esrc, const int* __restrict__ edst,
                          _Float16* __restrict__ ef0, int ne) {
  int t = blockIdx.x * 256 + threadIdx.x;
  if (t >= ne) return;
  int i = edst[t], j = esrc[t];
  const float* xi = x + (size_t)i * 7;
  const float* xj = x + (size_t)j * 7;
  float dpx = (xi[0] - xj[0]) * 20.0f;  // / RADIUS(0.05)
  float dpy = (xi[1] - xj[1]) * 20.0f;
  float r = sqrtf(dpx * dpx + dpy * dpy);
  int ti = (int)xi[6];
  float ai0 = atab[(size_t)ti * 2 + 0];
  _Float16* o = ef0 + (size_t)t * 32;
  o[0] = (_Float16)dpx; o[1] = (_Float16)dpy; o[2] = (_Float16)r;
  o[3] = (_Float16)xi[2]; o[4] = (_Float16)xi[3];
  o[5] = (_Float16)xj[2]; o[6] = (_Float16)xj[3];
#pragma unroll
  for (int k = 7; k < 11; ++k) o[k] = (_Float16)ai0;
#pragma unroll
  for (int k = 11; k < 32; ++k) o[k] = (_Float16)0.0f;
}

// ---------------------------------------------------------------- fused MLP kernels
// 8 waves / 256 threads per block, 128 rows per block, LDS-shared B panels.

template <int K1, int D1, int D2, int D3>
__global__ __launch_bounds__(256) void mlp3b_kernel(
    const _Float16* __restrict__ xin,
    const _Float16* __restrict__ w1t, const float* __restrict__ b1,
    const _Float16* __restrict__ w2t, const float* __restrict__ b2,
    const _Float16* __restrict__ w3t, const float* __restrict__ b3,
    _Float16* __restrict__ yout, int ntiles) {
  constexpr int LD1 = D1 + 8, LD2 = D2 + 8;
  constexpr int KM1 = (K1 > D1 ? K1 : D1);
  constexpr int KM = (KM1 > D2 ? KM1 : D2);
  constexpr int LDP = KM + 8;
  extern __shared__ _Float16 smem[];
  _Float16* actA = smem;                 // [128][LD1]
  _Float16* actB = actA + 128 * LD1;     // [128][LD2]
  _Float16* bpan = actB + 128 * LD2;     // [2][16*LDP]
  int wave = threadIdx.x >> 5;
  int tile = blockIdx.x * 8 + wave;
  bool on = tile < ntiles;
  const _Float16* xr = xin + (size_t)(on ? tile : 0) * 16 * K1;
  _Float16* h1w = actA + wave * 16 * LD1;
  _Float16* h2w = actB + wave * 16 * LD2;
  _Float16* yo = yout + (size_t)(on ? tile : 0) * 16 * D3;
  blayer<K1, D1>(xr, K1, w1t, b1, h1w, LD1, true, bpan, LDP, on);
  blayer<D1, D2>(h1w, LD1, w2t, b2, h2w, LD2, true, bpan, LDP, on);
  blayer<D2, D3>(h2w, LD2, w3t, b3, yo, D3, false, bpan, LDP, on);
}

template <int K1, int D1, int D2>
__global__ __launch_bounds__(256) void mlp2b_kernel(
    const _Float16* __restrict__ xin,
    const _Float16* __restrict__ w1t, const float* __restrict__ b1,
    const _Float16* __restrict__ w2t, const float* __restrict__ b2,
    _Float16* __restrict__ yout, int ntiles) {
  constexpr int LD1 = D1 + 8;
  constexpr int KM = (K1 > D1 ? K1 : D1);
  constexpr int LDP = KM + 8;
  extern __shared__ _Float16 smem[];
  _Float16* actA = smem;                 // [128][LD1]
  _Float16* bpan = actA + 128 * LD1;     // [2][16*LDP]
  int wave = threadIdx.x >> 5;
  int tile = blockIdx.x * 8 + wave;
  bool on = tile < ntiles;
  const _Float16* xr = xin + (size_t)(on ? tile : 0) * 16 * K1;
  _Float16* h1w = actA + wave * 16 * LD1;
  _Float16* yo = yout + (size_t)(on ? tile : 0) * 16 * D2;
  blayer<K1, D1>(xr, K1, w1t, b1, h1w, LD1, true, bpan, LDP, on);
  blayer<D1, D2>(h1w, LD1, w2t, b2, yo, D2, true, bpan, LDP, on);
}

// ---------------------------------------------------------------- message passing

__device__ __forceinline__ const _Float16* seg3(const _Float16* a, const _Float16* b,
                                                const _Float16* c, int col) {
  return col < 128 ? a + col : (col < 256 ? b + (col - 128) : c + (col - 256));
}

// per-edge: m = MLP_384_384_128([ef, nf[dst], nf[src]]); aggr[dst]+=m; ef+=m
__global__ __launch_bounds__(256) void edge_msg_kernel(
    const _Float16* __restrict__ nf16, _Float16* __restrict__ ef16,
    float* __restrict__ aggr,
    const int* __restrict__ esrc, const int* __restrict__ edst,
    const _Float16* __restrict__ w1t, const float* __restrict__ b1,
    const _Float16* __restrict__ w2t, const float* __restrict__ b2,
    const _Float16* __restrict__ w3t, const float* __restrict__ b3,
    int ntiles) {
  constexpr int LDA = 392, LDP = 392;
  extern __shared__ _Float16 smem[];
  _Float16* actA = smem;                 // [128][LDA] hidden1
  _Float16* actB = actA + 128 * LDA;     // [128][LDA] hidden2
  _Float16* bpan = actB + 128 * LDA;     // [2][16*LDP]
  __shared__ int sdst[128];

  const int wave = threadIdx.x >> 5;
  const int l = threadIdx.x & 31;
  const int tile = blockIdx.x * 8 + wave;
  const bool on = tile < ntiles;
  const int e0 = (on ? tile : 0) * 16;
  const int ei = e0 + (l & 15);
  const int di = edst[ei];
  const int si = esrc[ei];
  if (l < 16) sdst[wave * 16 + l] = di;
  const _Float16* efr = ef16 + (size_t)ei * 128;  // per-lane gathered rows
  const _Float16* nfd = nf16 + (size_t)di * 128;
  const _Float16* nfs = nf16 + (size_t)si * 128;
  _Float16* h1w = actA + wave * 16 * LDA;
  _Float16* h2w = actB + wave * 16 * LDA;

  // ---- layer 1: gathered concat(384) -> 384, relu
  v16h af[12];
  if (on) {
#pragma unroll
    for (int kk = 0; kk < 12; ++kk) {
      int k0 = kk * 32, kg = (l >> 4) * 8;
      union { v16h v; v8h h[2]; } u;
      u.h[0] = *(const v8h*)seg3(efr, nfd, nfs, k0 + kg);
      u.h[1] = *(const v8h*)seg3(efr, nfd, nfs, k0 + kg + 16);
      af[kk] = u.v;
    }
  }
  load_panel<384>(w1t, 0, bpan, LDP);
  __syncthreads();
  for (int n0 = 0; n0 < 384; n0 += 16) {
    const _Float16* cur = bpan + ((n0 >> 4) & 1) * (16 * LDP);
    _Float16* nxt = bpan + (((n0 >> 4) + 1) & 1) * (16 * LDP);
    if (n0 + 16 < 384) load_panel<384>(w1t, n0 + 16, nxt, LDP);
    if (on) {
      v8f acc = kloop<12>(af, cur, LDP);
      store_tile(acc, b1, n0, h1w, LDA, true);
    }
    __syncthreads();
  }
  // ---- layer 2: 384 -> 384, relu
  blayer<384, 384>(h1w, LDA, w2t, b2, h2w, LDA, true, bpan, LDP, on);
  // ---- layer 3: 384 -> 128, linear + scatter-add + ef residual
  if (on) {
#pragma unroll
    for (int kk = 0; kk < 12; ++kk) af[kk] = load_a(h2w, LDA, kk * 32);
  }
  load_panel<384>(w3t, 0, bpan, LDP);
  __syncthreads();
  for (int n0 = 0; n0 < 128; n0 += 16) {
    const _Float16* cur = bpan + ((n0 >> 4) & 1) * (16 * LDP);
    _Float16* nxt = bpan + (((n0 >> 4) + 1) & 1) * (16 * LDP);
    if (n0 + 16 < 128) load_panel<384>(w3t, n0 + 16, nxt, LDP);
    if (on) {
      v8f acc = kloop<12>(af, cur, LDP);
      int n = n0 + (l & 15);
      float bv = b3[n];
      int mb = (l >> 4) * 8;
#pragma unroll
      for (int r = 0; r < 8; ++r) {
        float v = acc[r] + bv;
        int row = mb + r;
        unsafeAtomicAdd(&aggr[(size_t)sdst[wave * 16 + row] * 128 + n], v);
        _Float16* pe = ef16 + (size_t)(e0 + row) * 128 + n;
        *pe = (_Float16)((float)*pe + v);
      }
    }
    __syncthreads();
  }
}

// per-node: nf += MLP_256_256_128([nf, aggr])
__global__ __launch_bounds__(256) void node_upd_kernel(
    _Float16* __restrict__ nf16, const float* __restrict__ aggr,
    const _Float16* __restrict__ w1t, const float* __restrict__ b1,
    const _Float16* __restrict__ w2t, const float* __restrict__ b2,
    const _Float16* __restrict__ w3t, const float* __restrict__ b3,
    int ntiles) {
  constexpr int LDA = 264, LDP = 264;
  extern __shared__ _Float16 smem[];
  _Float16* actA = smem;                 // [128][LDA] h1
  _Float16* actB = actA + 128 * LDA;     // [128][LDA] h2
  _Float16* bpan = actB + 128 * LDA;     // [2][16*LDP]
  const int wave = threadIdx.x >> 5;
  const int l = threadIdx.x & 31;
  const int tile = blockIdx.x * 8 + wave;
  const bool on = tile < ntiles;
  const int r0 = (on ? tile : 0) * 16;
  const _Float16* nfr = nf16 + (size_t)(r0 + (l & 15)) * 128;
  const float* agr = aggr + (size_t)(r0 + (l & 15)) * 128;
  _Float16* h1w = actA + wave * 16 * LDA;
  _Float16* h2w = actB + wave * 16 * LDA;

  // ---- layer 1: concat [nf(f16), aggr(f32->f16)] (256) -> 256, relu
  v16h af[8];
  if (on) {
#pragma unroll
    for (int kk = 0; kk < 8; ++kk) {
      int k0 = kk * 32, kg = (l >> 4) * 8;
      int c0 = k0 + kg, c1 = c0 + 16;
      union { v16h v; v8h h[2]; } u;
      u.h[0] = (c0 < 128) ? *(const v8h*)(nfr + c0) : cvt8(agr + (c0 - 128));
      u.h[1] = (c1 < 128) ? *(const v8h*)(nfr + c1) : cvt8(agr + (c1 - 128));
      af[kk] = u.v;
    }
  }
  load_panel<256>(w1t, 0, bpan, LDP);
  __syncthreads();
  for (int n0 = 0; n0 < 256; n0 += 16) {
    const _Float16* cur = bpan + ((n0 >> 4) & 1) * (16 * LDP);
    _Float16* nxt = bpan + (((n0 >> 4) + 1) & 1) * (16 * LDP);
    if (n0 + 16 < 256) load_panel<256>(w1t, n0 + 16, nxt, LDP);
    if (on) {
      v8f acc = kloop<8>(af, cur, LDP);
      store_tile(acc, b1, n0, h1w, LDA, true);
    }
    __syncthreads();
  }
  // ---- layer 2
  blayer<256, 256>(h1w, LDA, w2t, b2, h2w, LDA, true, bpan, LDP, on);
  // ---- layer 3 + residual into nf
  if (on) {
#pragma unroll
    for (int kk = 0; kk < 8; ++kk) af[kk] = load_a(h2w, LDA, kk * 32);
  }
  load_panel<256>(w3t, 0, bpan, LDP);
  __syncthreads();
  for (int n0 = 0; n0 < 128; n0 += 16) {
    const _Float16* cur = bpan + ((n0 >> 4) & 1) * (16 * LDP);
    _Float16* nxt = bpan + (((n0 >> 4) + 1) & 1) * (16 * LDP);
    if (n0 + 16 < 128) load_panel<256>(w3t, n0 + 16, nxt, LDP);
    if (on) {
      v8f acc = kloop<8>(af, cur, LDP);
      int n = n0 + (l & 15);
      float bv = b3[n];
      int mb = (l >> 4) * 8;
#pragma unroll
      for (int r = 0; r < 8; ++r) {
        _Float16* p = nf16 + (size_t)(r0 + mb + r) * 128 + n;
        *p = (_Float16)((float)*p + acc[r] + bv);
      }
    }
    __syncthreads();
  }
}

// final tiny layer: out[n][0:2] = h2[n] @ W3(128x2) + b3
__global__ void node_out_final(const _Float16* __restrict__ h2,
                               const float* __restrict__ w3, const float* __restrict__ b3,
                               float* __restrict__ out, int n) {
  int t = blockIdx.x * 256 + threadIdx.x;
  if (t >= n) return;
  const _Float16* hr = h2 + (size_t)t * 128;
  float a0 = b3[0], a1 = b3[1];
#pragma unroll 4
  for (int k = 0; k < 128; ++k) {
    float h = (float)hr[k];
    a0 += h * w3[k * 2 + 0];
    a1 += h * w3[k * 2 + 1];
  }
  out[t * 2 + 0] = a0;
  out[t * 2 + 1] = a1;
}

// ---------------------------------------------------------------- launch
extern "C" void kernel_launch(void* const* d_in, const int* in_sizes, int n_in,
                              void* d_out, int out_size, void* d_ws, size_t ws_size,
                              hipStream_t stream) {
  (void)in_sizes; (void)n_in; (void)out_size; (void)ws_size;
  const float* x    = (const float*)d_in[0];
  const float* atab = (const float*)d_in[1];
  const float* enW1 = (const float*)d_in[2];  const float* enB1 = (const float*)d_in[3];
  const float* enW2 = (const float*)d_in[4];  const float* enB2 = (const float*)d_in[5];
  const float* enW3 = (const float*)d_in[6];  const float* enB3 = (const float*)d_in[7];
  const float* eeW1 = (const float*)d_in[8];  const float* eeB1 = (const float*)d_in[9];
  const float* eeW2 = (const float*)d_in[10]; const float* eeB2 = (const float*)d_in[11];
  const float* eeW3 = (const float*)d_in[12]; const float* eeB3 = (const float*)d_in[13];
  const float* leW1 = (const float*)d_in[14]; const float* leB1 = (const float*)d_in[15];
  const float* leW2 = (const float*)d_in[16]; const float* leB2 = (const float*)d_in[17];
  const float* leW3 = (const float*)d_in[18]; const float* leB3 = (const float*)d_in[19];
  const float* lnW1 = (const float*)d_in[20]; const float* lnB1 = (const float*)d_in[21];
  const float* lnW2 = (const float*)d_in[22]; const float* lnB2 = (const float*)d_in[23];
  const float* lnW3 = (const float*)d_in[24]; const float* lnB3 = (const float*)d_in[25];
  const float* noW1 = (const float*)d_in[26]; const float* noB1 = (const float*)d_in[27];
  const float* noW2 = (const float*)d_in[28]; const float* noB2 = (const float*)d_in[29];
  const float* noW3 = (const float*)d_in[30]; const float* noB3 = (const float*)d_in[31];
  const int*   eidx = (const int*)d_in[32];
  const int* esrc = eidx;       // edge_index row 0 = src (j)
  const int* edst = eidx + NE;  // edge_index row 1 = dst (i)

  char* ws = (char*)d_ws;
  size_t off = 0;
  auto alloc = [&](size_t bytes) -> char* {
    char* p = ws + off;
    off += (bytes + 255) & ~(size_t)255;
    return p;
  };
  _Float16* nf0  = (_Float16*)alloc((size_t)NN * 32 * 2);
  _Float16* ef0  = (_Float16*)alloc((size_t)NE * 32 * 2);
  _Float16* nf16 = (_Float16*)alloc((size_t)NN * 128 * 2);
  _Float16* ef16 = (_Float16*)alloc((size_t)NE * 128 * 2);
  float*    aggr = (float*)   alloc((size_t)NN * 128 * 4);
  _Float16* h2t  = (_Float16*)alloc((size_t)NN * 128 * 2);
  _Float16* wEN1 = (_Float16*)alloc(128 * 32 * 2);
  _Float16* wEN2 = (_Float16*)alloc(128 * 128 * 2);
  _Float16* wEN3 = (_Float16*)alloc(128 * 128 * 2);
  _Float16* wEE1 = (_Float16*)alloc(128 * 32 * 2);
  _Float16* wEE2 = (_Float16*)alloc(128 * 128 * 2);
  _Float16* wEE3 = (_Float16*)alloc(128 * 128 * 2);
  _Float16* wLE1 = (_Float16*)alloc(384 * 384 * 2);
  _Float16* wLE2 = (_Float16*)alloc(384 * 384 * 2);
  _Float16* wLE3 = (_Float16*)alloc(128 * 384 * 2);
  _Float16* wLN1 = (_Float16*)alloc(256 * 256 * 2);
  _Float16* wLN2 = (_Float16*)alloc(256 * 256 * 2);
  _Float16* wLN3 = (_Float16*)alloc(128 * 256 * 2);
  _Float16* wNO1 = (_Float16*)alloc(128 * 128 * 2);
  _Float16* wNO2 = (_Float16*)alloc(128 * 128 * 2);

  auto castw = [&](const float* s, _Float16* d, int din, int dout, int kpad) {
    int tot = dout * kpad;
    wcast_kernel<<<(tot + 255) / 256, 256, 0, stream>>>(s, d, din, dout, kpad);
  };
  castw(enW1, wEN1, 12, 128, 32);   castw(enW2, wEN2, 128, 128, 128); castw(enW3, wEN3, 128, 128, 128);
  castw(eeW1, wEE1, 11, 128, 32);   castw(eeW2, wEE2, 128, 128, 128); castw(eeW3, wEE3, 128, 128, 128);
  castw(leW1, wLE1, 384, 384, 384); castw(leW2, wLE2, 384, 384, 384); castw(leW3, wLE3, 384, 128, 384);
  castw(lnW1, wLN1, 256, 256, 256); castw(lnW2, wLN2, 256, 256, 256); castw(lnW3, wLN3, 256, 128, 256);
  castw(noW1, wNO1, 128, 128, 128); castw(noW2, wNO2, 128, 128, 128);

  build_nf0<<<(NN + 255) / 256, 256, 0, stream>>>(x, atab, nf0, NN);
  build_ef0<<<(NE + 255) / 256, 256, 0, stream>>>(x, atab, esrc, edst, ef0, NE);

  const int ntN = NN / 16, ntE = NE / 16;  // 1250 / 20000 tiles
  const size_t smMLP3 = (size_t)(2 * 128 * 136 + 2 * 16 * 136) * 2;  // 78,336 B
  const size_t smMLP2 = (size_t)(128 * 136 + 2 * 16 * 136) * 2;      // 43,520 B
  const size_t smEDGE = (size_t)(2 * 128 * 392 + 2 * 16 * 392) * 2;  // 225,792 B
  const size_t smNODE = (size_t)(2 * 128 * 264 + 2 * 16 * 264) * 2;  // 152,064 B

  mlp3b_kernel<32, 128, 128, 128><<<(ntN + 7) / 8, 256, smMLP3, stream>>>(
      nf0, wEN1, enB1, wEN2, enB2, wEN3, enB3, nf16, ntN);
  mlp3b_kernel<32, 128, 128, 128><<<(ntE + 7) / 8, 256, smMLP3, stream>>>(
      ef0, wEE1, eeB1, wEE2, eeB2, wEE3, eeB3, ef16, ntE);

  for (int it = 0; it < 4; ++it) {
    zero_f32<<<(NN * 128 + 255) / 256, 256, 0, stream>>>(aggr, NN * 128);
    edge_msg_kernel<<<(ntE + 7) / 8, 256, smEDGE, stream>>>(
        nf16, ef16, aggr, esrc, edst, wLE1, leB1, wLE2, leB2, wLE3, leB3, ntE);
    node_upd_kernel<<<(ntN + 7) / 8, 256, smNODE, stream>>>(
        nf16, aggr, wLN1, lnB1, wLN2, lnB2, wLN3, lnB3, ntN);
  }

  mlp2b_kernel<128, 128, 128><<<(ntN + 7) / 8, 256, smMLP2, stream>>>(
      nf16, wNO1, noB1, wNO2, noB2, h2t, ntN);
  node_out_final<<<(NN + 255) / 256, 256, 0, stream>>>(h2t, noW3, noB3,
                                                       (float*)d_out, NN);
}
